// MultiHeadAttention_12446815224567
// MI455X (gfx1250) — matrix operand order repaired and checked by reading, not compiled, not run on hardware
//
#include <hip/hip_runtime.h>

// ---------------------------------------------------------------------------
// MI455X (gfx1250) multi-head attention: bf16 WMMA pipeline, f32 accumulate.
//   EMBED=768, HEADS=12, HEAD_DIM=64, B=8, P=1024
// ---------------------------------------------------------------------------

typedef __bf16 bf16_t;
typedef __attribute__((ext_vector_type(16))) __bf16 v16bf;
typedef __attribute__((ext_vector_type(8)))  __bf16 v8bf;
typedef __attribute__((ext_vector_type(8)))  float  v8f;

#define EMBED   768
#define QKVDIM  2304
#define HEADS   12
#define HDIM    64
#define BATCH   8
#define SEQ     1024
#define ROWS    (BATCH * SEQ)          // 8192

// XOR-butterfly over 16 lanes via ds_swizzle (group-of-32: offset=(xor<<10)|0x1f)
#define XOR_SWZ(x, imm) \
    __builtin_bit_cast(float, __builtin_amdgcn_ds_swizzle(__builtin_bit_cast(int, (x)), (imm)))

__device__ __forceinline__ float redmax16(float x) {
    x = fmaxf(x, XOR_SWZ(x, 0x041F));
    x = fmaxf(x, XOR_SWZ(x, 0x081F));
    x = fmaxf(x, XOR_SWZ(x, 0x101F));
    x = fmaxf(x, XOR_SWZ(x, 0x201F));
    return x;
}
__device__ __forceinline__ float redsum16(float x) {
    x += XOR_SWZ(x, 0x041F);
    x += XOR_SWZ(x, 0x081F);
    x += XOR_SWZ(x, 0x101F);
    x += XOR_SWZ(x, 0x201F);
    return x;
}

// ---------------- WMMA fragment loaders (ISA 7.12.2 layouts) ----------------

__device__ __forceinline__ v16bf concat8(v8bf c0, v8bf c1) {
    return __builtin_shufflevector(c0, c1, 0, 1, 2, 3, 4, 5, 6, 7,
                                           8, 9, 10, 11, 12, 13, 14, 15);
}

// A-matrix 16x32 bf16 (M x K), row-major source, leading dim lda.
// lane 0-15: M=lane, K chunks {kb.., kb+16..}; lane 16-31: M=lane-16, +8.
__device__ __forceinline__ v16bf load_a_frag(const bf16_t* base, int lda,
                                             int mrow, int kb) {
    const int lane = threadIdx.x & 31;
    const int row  = mrow + (lane & 15);
    const int k0   = kb + ((lane >> 4) << 3);
    const bf16_t* p = base + (size_t)row * lda + k0;
    return concat8(*(const v8bf*)(p), *(const v8bf*)(p + 16));
}

// B-matrix 32x16 bf16 (K x N), source N-major: element (k,n) at base[n*ldb+k].
// lane 0-15: column N=lane, K chunks {kb.., kb+16..}; lane 16-31: N=lane-16, +8.
__device__ __forceinline__ v16bf load_b_frag(const bf16_t* base, int ldb,
                                             int ncol, int kb) {
    const int lane = threadIdx.x & 31;
    const int col  = ncol + (lane & 15);
    const int k0   = kb + ((lane >> 4) << 3);
    const bf16_t* p = base + (size_t)col * ldb + k0;
    return concat8(*(const v8bf*)(p), *(const v8bf*)(p + 16));
}

__device__ __forceinline__ v8f wmma_bf16(v16bf a, v16bf b, v8f c) {
    return __builtin_amdgcn_wmma_f32_16x16x32_bf16(
        false, a, false, b, (short)0, c, false, false);
}

// ---------------------------- f32 -> bf16 convert ---------------------------
// 8 elements per thread: 2x b128 loads -> 1x b128 store.

__global__ void mha_cvt_bf16(const float* __restrict__ in,
                             bf16_t* __restrict__ out, int n8) {
    int i = blockIdx.x * blockDim.x + threadIdx.x;
    const int stride = gridDim.x * blockDim.x;
    for (; i < n8; i += stride) {
        const float4 a = ((const float4*)in)[2 * i];
        const float4 b = ((const float4*)in)[2 * i + 1];
        v8bf o;
        o[0] = (bf16_t)a.x; o[1] = (bf16_t)a.y; o[2] = (bf16_t)a.z; o[3] = (bf16_t)a.w;
        o[4] = (bf16_t)b.x; o[5] = (bf16_t)b.y; o[6] = (bf16_t)b.z; o[7] = (bf16_t)b.w;
        ((v8bf*)out)[i] = o;
    }
}

// ------------------------------- tiled GEMM ---------------------------------
// C[M][N] = A[M][K] * W[N][K]^T + bias[N]
// 256 threads = 8 waves, 2(M) x 4(N); wave tile 64x64; block tile 128x256.

template <bool OUT_BF16, bool SCALE_Q>
__global__ __launch_bounds__(256)
void mha_gemm(const bf16_t* __restrict__ A, const bf16_t* __restrict__ W,
              const float* __restrict__ bias, void* __restrict__ Cout,
              int K, int ldc) {
    const int wave  = threadIdx.x >> 5;
    const int lane  = threadIdx.x & 31;
    const int nlane = lane & 15;
    const int half  = lane >> 4;
    const int mbase = blockIdx.x * 128 + (wave >> 2) * 64;
    const int nbase = blockIdx.y * 256 + (wave & 3) * 64;

    v8f acc[4][4] = {};

    for (int kb = 0; kb < K; kb += 32) {
        if (kb + 32 < K) {  // global_prefetch_b8 hints for next K-slab
            __builtin_prefetch(A + (size_t)(mbase + nlane) * K + kb + 32, 0, 3);
            __builtin_prefetch(W + (size_t)(nbase + nlane) * K + kb + 32, 0, 3);
        }
        v16bf af[4], bfr[4];
#pragma unroll
        for (int i = 0; i < 4; ++i) af[i]  = load_a_frag(A, K, mbase + 16 * i, kb);
#pragma unroll
        for (int j = 0; j < 4; ++j) bfr[j] = load_b_frag(W, K, nbase + 16 * j, kb);
#pragma unroll
        for (int i = 0; i < 4; ++i)
#pragma unroll
            for (int j = 0; j < 4; ++j)
                acc[i][j] = wmma_bf16(af[i], bfr[j], acc[i][j]);
    }

#pragma unroll
    for (int i = 0; i < 4; ++i) {
#pragma unroll
        for (int j = 0; j < 4; ++j) {
            const int col = nbase + j * 16 + nlane;
            const float b = bias[col];
#pragma unroll
            for (int r = 0; r < 8; ++r) {
                const int row = mbase + i * 16 + half * 8 + r;
                float v = acc[i][j][r] + b;
                if (SCALE_Q) { if (col < EMBED) v *= 0.125f; }  // HEAD_DIM^-0.5
                if (OUT_BF16)
                    ((bf16_t*)Cout)[(size_t)row * ldc + col] = (bf16_t)v;
                else
                    ((float*)Cout)[(size_t)row * ldc + col] = v;
            }
        }
    }
}

// ----------------------- flash attention (per b,h) --------------------------
// qkvbf: [8192][2304] bf16; cols [0,768)=Q(prescaled), [768,1536)=K,
// [1536,2304)=V. ctx: [8192][768] bf16.
// Grid: 96 (b*h) * 16 query blocks. Block: 128 threads = 4 waves x 16 queries.

#define VT_PAD 40   // 80B row stride: 16B aligned, bank-conflict free

__global__ __launch_bounds__(128)
void mha_attn(const bf16_t* __restrict__ qkv, bf16_t* __restrict__ ctx) {
    const int bh    = blockIdx.x >> 4;
    const int qblk  = blockIdx.x & 15;
    const int b     = bh / HEADS;
    const int h     = bh % HEADS;
    const int wave  = threadIdx.x >> 5;
    const int lane  = threadIdx.x & 31;
    const int nlane = lane & 15;
    const int half  = lane >> 4;

    __shared__ bf16_t Vt[HDIM][VT_PAD];      // transposed V block: [d][key]
    __shared__ bf16_t Plds[4][16][VT_PAD];   // per-wave probability tile 16x32

    const bf16_t* base = qkv + (size_t)(b * SEQ) * QKVDIM;
    const int hq = h * HDIM;
    const int hk = EMBED + h * HDIM;
    const int hv = 2 * EMBED + h * HDIM;
    const int q0 = qblk * 64 + wave * 16;

    // Q fragments (A layout); two K-steps cover d=0..63. Q pre-scaled.
    v16bf qf[2];
#pragma unroll
    for (int ks = 0; ks < 2; ++ks) {
        const bf16_t* p = base + (size_t)(q0 + nlane) * QKVDIM + hq
                        + ks * 32 + half * 8;
        qf[ks] = concat8(*(const v8bf*)(p), *(const v8bf*)(p + 16));
    }

    float m_run[8], l_run[8];
#pragma unroll
    for (int r = 0; r < 8; ++r) { m_run[r] = -__builtin_inff(); l_run[r] = 0.0f; }
    v8f o[4] = {};

    for (int kb = 0; kb < SEQ; kb += 32) {
        // Stage V[kb..kb+31][0..63] transposed into LDS: 16B global loads,
        // bf16 scatter on the LDS side only.
#pragma unroll
        for (int it = 0; it < 2; ++it) {
            const int idx = threadIdx.x + it * 128;  // 0..255
            const int key = idx >> 3;                // 0..31
            const int dc  = (idx & 7) * 8;           // 0,8,..,56
            v8bf v = *(const v8bf*)&base[(size_t)(kb + key) * QKVDIM + hv + dc];
#pragma unroll
            for (int j = 0; j < 8; ++j) Vt[dc + j][key] = v[j];
        }
        __syncthreads();

        // K fragments first (one load clause), then the 4 score WMMAs.
        v16bf kf[2][2];
#pragma unroll
        for (int t = 0; t < 2; ++t)
#pragma unroll
            for (int ks = 0; ks < 2; ++ks)
                kf[t][ks] = load_b_frag(base + hk, QKVDIM, kb + t * 16, ks * 32);

        // V fragments early: only depend on Vt + barrier; overlap softmax.
        v16bf vf[4];
#pragma unroll
        for (int t2 = 0; t2 < 4; ++t2)
            vf[t2] = concat8(*(const v8bf*)&Vt[t2 * 16 + nlane][half * 8],
                             *(const v8bf*)&Vt[t2 * 16 + nlane][half * 8 + 16]);

        v8f s[2] = {};
#pragma unroll
        for (int t = 0; t < 2; ++t)
#pragma unroll
            for (int ks = 0; ks < 2; ++ks)
                s[t] = wmma_bf16(qf[ks], kf[t][ks], s[t]);

        // Online softmax over this 32-key block.
#pragma unroll
        for (int r = 0; r < 8; ++r) {
            const float v0 = s[0][r], v1 = s[1][r];
            const float mx   = redmax16(fmaxf(v0, v1));
            const float mnew = fmaxf(m_run[r], mx);
            const float p0 = __expf(v0 - mnew);
            const float p1 = __expf(v1 - mnew);
            const float rs = redsum16(p0 + p1);
            const float corr = __expf(m_run[r] - mnew);
            l_run[r] = l_run[r] * corr + rs;
            m_run[r] = mnew;
#pragma unroll
            for (int t2 = 0; t2 < 4; ++t2) o[t2][r] *= corr;
            const int prow = half * 8 + r;           // C-layout -> row-major LDS
            Plds[wave][prow][nlane]      = (bf16_t)p0;
            Plds[wave][prow][16 + nlane] = (bf16_t)p1;
        }

        // Reload P as an A fragment (16 x 32 over keys), wave-private.
        const bf16_t* pp = &Plds[wave][nlane][half * 8];
        v16bf pf = concat8(*(const v8bf*)(pp), *(const v8bf*)(pp + 16));

        // O[16q x 64d] += P(16x32) * V(32x64)
#pragma unroll
        for (int t2 = 0; t2 < 4; ++t2)
            o[t2] = wmma_bf16(pf, vf[t2], o[t2]);

        __syncthreads();   // protect Vt before the next block overwrites it
    }

    // Finalize: divide by row sums, store context bf16 [8192][768].
#pragma unroll
    for (int r = 0; r < 8; ++r) {
        const float inv = 1.0f / l_run[r];
        const size_t grow = (size_t)(b * SEQ + q0 + half * 8 + r);
#pragma unroll
        for (int t2 = 0; t2 < 4; ++t2)
            ctx[grow * EMBED + h * HDIM + t2 * 16 + nlane] =
                (bf16_t)(o[t2][r] * inv);
    }
}

// ------------------------------ host launcher -------------------------------

extern "C" void kernel_launch(void* const* d_in, const int* in_sizes, int n_in,
                              void* d_out, int out_size, void* d_ws, size_t ws_size,
                              hipStream_t stream) {
    (void)in_sizes; (void)n_in; (void)out_size; (void)ws_size;

    const float* x_q   = (const float*)d_in[0];   // [8,1024,768]
    const float* qkv_w = (const float*)d_in[1];   // [2304,768]
    const float* qkv_b = (const float*)d_in[2];   // [2304]
    const float* out_w = (const float*)d_in[3];   // [768,768]
    const float* out_b = (const float*)d_in[4];   // [768]
    float* out = (float*)d_out;                   // [8,1024,768]

    char* ws = (char*)d_ws;
    bf16_t* Xbf   = (bf16_t*)(ws);                                   // 12.58 MB
    bf16_t* Wqkv  = (bf16_t*)(ws + 12582912);                        //  3.54 MB
    bf16_t* Wout  = (bf16_t*)(ws + 12582912 + 3538944);              //  1.18 MB
    bf16_t* QKVbf = (bf16_t*)(ws + 12582912 + 3538944 + 1179648);    // 37.75 MB
    bf16_t* CTX   = (bf16_t*)(ws + 12582912 + 3538944 + 1179648 + 37748736);

    mha_cvt_bf16<<<1024, 256, 0, stream>>>(x_q,   Xbf,  (ROWS * EMBED) / 8);
    mha_cvt_bf16<<<512,  256, 0, stream>>>(qkv_w, Wqkv, (QKVDIM * EMBED) / 8);
    mha_cvt_bf16<<<288,  256, 0, stream>>>(out_w, Wout, (EMBED * EMBED) / 8);

    mha_gemm<true, true><<<dim3(ROWS / 128, QKVDIM / 256), 256, 0, stream>>>(
        Xbf, Wqkv, qkv_b, (void*)QKVbf, EMBED, QKVDIM);

    mha_attn<<<BATCH * HEADS * (SEQ / 64), 128, 0, stream>>>(QKVbf, CTX);

    mha_gemm<false, false><<<dim3(ROWS / 128, EMBED / 256), 256, 0, stream>>>(
        CTX, Wout, out_b, (void*)out, EMBED, EMBED);
}
